// TrmBlock_36472862277693
// MI455X (gfx1250) — matrix operand order repaired and checked by reading, not compiled, not run on hardware
//
#include <hip/hip_runtime.h>
#include <hip/hip_bf16.h>

// ---------------------------------------------------------------------------
// Transformer block for MI455X (gfx1250, wave32, WMMA bf16).
//   B=16, N=1024, D=512, P=512, H=8, E=64, F=4*D=2048, M=B*N=16384
// All matmuls in bf16 via v_wmma_f32_16x16x32_bf16 with f32 accumulation.
// ---------------------------------------------------------------------------

typedef __bf16 bf16;
typedef __attribute__((ext_vector_type(16))) __bf16 v16bf;
typedef __attribute__((ext_vector_type(8)))  float  v8f;

constexpr int B_ = 16;
constexpr int N_ = 1024;
constexpr int D_ = 512;
constexpr int P_ = 512;
constexpr int H_ = 8;
constexpr int E_ = 64;
constexpr int F_ = 2048;
constexpr int M_ = B_ * N_;   // 16384 tokens

__device__ inline v8f v8zero() {
    v8f z;
#pragma unroll
    for (int i = 0; i < 8; ++i) z[i] = 0.0f;
    return z;
}

union FragU { v16bf v; uint4 q[2]; };

// A-matrix 16x32 bf16 frag: lane<16 -> K {0..7, 16..23}, lane>=16 -> K {8..15, 24..31}
__device__ inline v16bf load_a_frag(const bf16* p, int hf) {
    FragU f;
    f.q[0] = *(const uint4*)(p + hf * 8);
    f.q[1] = *(const uint4*)(p + hf * 8 + 16);
    return f.v;
}
// B-matrix 32x16 bf16 frag: lane = column, lane<16 -> K {0..7, 16..23}, lane>=16 -> K {8..15, 24..31}
__device__ inline v16bf load_b_frag(const bf16* p, int hf) {
    FragU f;
    f.q[0] = *(const uint4*)(p + hf * 8);
    f.q[1] = *(const uint4*)(p + hf * 8 + 16);
    return f.v;
}

__device__ inline v8f wmma_bf16(v16bf a, v16bf b, v8f c) {
    return __builtin_amdgcn_wmma_f32_16x16x32_bf16(
        false, a, false, b, (short)0, c, false, false);
}

__device__ inline float gelu_exact(float x) {
    return 0.5f * x * (1.0f + erff(x * 0.70710678118654752f));
}

// ---------------------------------------------------------------------------
// Generic GEMM: out[M,Nd] = A[M,K](bf16) x Wt[Nd,K](bf16, pre-transposed)
//               + bias (+ resid) (+ GELU). Dual f32/bf16 output.
// Block = 128 threads (4 waves); wave tile = 32(M) x 64(N); block = 128 x 64.
// ---------------------------------------------------------------------------
__global__ __launch_bounds__(128) void gemm_bf16_kernel(
    const bf16* __restrict__ A, const bf16* __restrict__ Wt,
    const float* __restrict__ bias, const float* __restrict__ resid,
    float* __restrict__ outF, bf16* __restrict__ outB,
    int Mdim, int Ndim, int Kdim, int act)
{
    const int tid  = threadIdx.x;
    const int wave = tid >> 5;
    const int lane = tid & 31;
    const int hf   = lane >> 4;
    const int l15  = lane & 15;

    const int m0 = blockIdx.y * 128 + wave * 32;
    const int n0 = blockIdx.x * 64;

    v8f acc[2][4];
#pragma unroll
    for (int mi = 0; mi < 2; ++mi)
#pragma unroll
        for (int j = 0; j < 4; ++j) acc[mi][j] = v8zero();

    const bf16* arow0 = A + (size_t)(m0 + l15) * Kdim;
    const bf16* arow1 = A + (size_t)(m0 + 16 + l15) * Kdim;

    for (int k = 0; k < Kdim; k += 32) {
        v16bf a0 = load_a_frag(arow0 + k, hf);
        v16bf a1 = load_a_frag(arow1 + k, hf);
#pragma unroll
        for (int j = 0; j < 4; ++j) {
            const bf16* wcol = Wt + (size_t)(n0 + j * 16 + l15) * Kdim + k;
            v16bf b = load_b_frag(wcol, hf);
            acc[0][j] = wmma_bf16(a0, b, acc[0][j]);
            acc[1][j] = wmma_bf16(a1, b, acc[1][j]);
        }
    }

#pragma unroll
    for (int mi = 0; mi < 2; ++mi) {
#pragma unroll
        for (int j = 0; j < 4; ++j) {
            const int col = n0 + j * 16 + l15;
            const float bv = bias[col];
#pragma unroll
            for (int r = 0; r < 8; ++r) {
                const int row = m0 + mi * 16 + r + 8 * hf;
                float v = acc[mi][j][r] + bv;
                if (resid) v += resid[(size_t)row * Ndim + col];
                if (act == 1) v = gelu_exact(v);
                const size_t oidx = (size_t)row * Ndim + col;
                if (outF) outF[oidx] = v;
                if (outB) outB[oidx] = (bf16)v;
            }
        }
    }
}

// ---------------------------------------------------------------------------
// Fused flash attention. q,k: [M,P] bf16; vt: [B,H,E,N] bf16; vout: [M,P] bf16.
// Block = 128 threads (4 waves); wave handles 16 q-rows of one (b,h);
// block handles 64 consecutive q-rows. Online softmax, scale = 1/sqrt(E).
// ---------------------------------------------------------------------------
__global__ __launch_bounds__(128) void attn_kernel(
    const bf16* __restrict__ q, const bf16* __restrict__ k,
    const bf16* __restrict__ vt, bf16* __restrict__ vout)
{
    __shared__ bf16 psh[4][16][32];   // per-wave 16x32 probability tile

    const int tid  = threadIdx.x;
    const int wave = tid >> 5;
    const int lane = tid & 31;
    const int hf   = lane >> 4;
    const int l15  = lane & 15;

    const int tiles = N_ / 64;
    const int bh   = blockIdx.x / tiles;
    const int tile = blockIdx.x % tiles;
    const int b = bh / H_, h = bh % H_;
    const int rowBase = tile * 64 + wave * 16;

    const bf16* qrow = q + ((size_t)(b * N_ + rowBase + l15)) * P_ + h * E_;
    const v16bf qa0 = load_a_frag(qrow, hf);        // e 0..31
    const v16bf qa1 = load_a_frag(qrow + 32, hf);   // e 32..63

    float mr[8], lr[8];
#pragma unroll
    for (int r = 0; r < 8; ++r) { mr[r] = -3.0e38f; lr[r] = 0.0f; }
    v8f o[4];
#pragma unroll
    for (int s = 0; s < 4; ++s) o[s] = v8zero();

    const bf16* vbase = vt + ((size_t)((b * H_ + h) * E_)) * N_;

    for (int s0 = 0; s0 < N_; s0 += 32) {
        // scores: q(16x64) x k^T(64x32) -> two 16x16 tiles
        const bf16* kp0 = k + ((size_t)(b * N_ + s0 + l15)) * P_ + h * E_;
        const bf16* kp1 = kp0 + (size_t)16 * P_;
        v8f c0 = wmma_bf16(qa0, load_b_frag(kp0, hf), v8zero());
        c0     = wmma_bf16(qa1, load_b_frag(kp0 + 32, hf), c0);
        v8f c1 = wmma_bf16(qa0, load_b_frag(kp1, hf), v8zero());
        c1     = wmma_bf16(qa1, load_b_frag(kp1 + 32, hf), c1);

#pragma unroll
        for (int r = 0; r < 8; ++r) {
            float x0 = c0[r] * 0.125f;   // 1/sqrt(64)
            float x1 = c1[r] * 0.125f;
            float t = fmaxf(x0, x1);
            t = fmaxf(t, __shfl_xor(t, 1, 32));
            t = fmaxf(t, __shfl_xor(t, 2, 32));
            t = fmaxf(t, __shfl_xor(t, 4, 32));
            t = fmaxf(t, __shfl_xor(t, 8, 32));
            float mnew = fmaxf(mr[r], t);
            float corr = __expf(mr[r] - mnew);
            mr[r] = mnew;
            float p0 = __expf(x0 - mnew);
            float p1 = __expf(x1 - mnew);
            float su = p0 + p1;
            su += __shfl_xor(su, 1, 32);
            su += __shfl_xor(su, 2, 32);
            su += __shfl_xor(su, 4, 32);
            su += __shfl_xor(su, 8, 32);
            lr[r] = lr[r] * corr + su;
#pragma unroll
            for (int s = 0; s < 4; ++s) o[s][r] *= corr;
            psh[wave][r + 8 * hf][l15]      = (bf16)p0;
            psh[wave][r + 8 * hf][16 + l15] = (bf16)p1;
        }

        asm volatile("s_wait_dscnt 0" ::: "memory");  // C-layout -> A-layout via LDS

        v16bf pa = load_a_frag(&psh[wave][l15][0], hf);
#pragma unroll
        for (int s = 0; s < 4; ++s) {
            const bf16* vp = vbase + (size_t)(s * 16 + l15) * N_ + s0;
            o[s] = wmma_bf16(pa, load_b_frag(vp, hf), o[s]);
        }
        asm volatile("s_wait_dscnt 0" ::: "memory");  // reads done before next overwrite
    }

#pragma unroll
    for (int r = 0; r < 8; ++r) mr[r] = 1.0f / lr[r];
#pragma unroll
    for (int s = 0; s < 4; ++s) {
#pragma unroll
        for (int r = 0; r < 8; ++r) {
            const int row = rowBase + r + 8 * hf;
            vout[((size_t)(b * N_ + row)) * P_ + h * E_ + s * 16 + l15] =
                (bf16)(o[s][r] * mr[r]);
        }
    }
}

// ---------------------------------------------------------------------------
// LayerNorm: one wave per token (D=512 -> 16 floats/lane), shuffle reductions.
// ---------------------------------------------------------------------------
__global__ __launch_bounds__(256) void ln_kernel(
    const float* __restrict__ in, const float* __restrict__ g,
    const float* __restrict__ bb, float* __restrict__ outF,
    bf16* __restrict__ outB)
{
    const int wave = threadIdx.x >> 5;
    const int lane = threadIdx.x & 31;
    const int token = blockIdx.x * 8 + wave;
    const float* row = in + (size_t)token * D_;

    float4 x[4];
    float s = 0.0f, ss = 0.0f;
#pragma unroll
    for (int i = 0; i < 4; ++i) {
        x[i] = ((const float4*)row)[lane * 4 + i];
        s  += x[i].x + x[i].y + x[i].z + x[i].w;
        ss += x[i].x * x[i].x + x[i].y * x[i].y + x[i].z * x[i].z + x[i].w * x[i].w;
    }
#pragma unroll
    for (int off = 1; off < 32; off <<= 1) {
        s  += __shfl_xor(s,  off, 32);
        ss += __shfl_xor(ss, off, 32);
    }
    const float mu  = s * (1.0f / D_);
    const float var = ss * (1.0f / D_) - mu * mu;
    const float rs  = rsqrtf(var + 1e-5f);

    const float* xf = (const float*)x;
    const int e0 = lane * 16;
#pragma unroll
    for (int i = 0; i < 4; ++i) {
        float4 ov;
        float* ovf = (float*)&ov;
#pragma unroll
        for (int c = 0; c < 4; ++c) {
            const int e = e0 + i * 4 + c;
            ovf[c] = (xf[i * 4 + c] - mu) * rs * g[e] + bb[e];
        }
        ((float4*)(outF + (size_t)token * D_))[lane * 4 + i] = ov;
        if (outB) {
#pragma unroll
            for (int c = 0; c < 4; ++c)
                outB[(size_t)token * D_ + e0 + i * 4 + c] = (bf16)ovf[c];
        }
    }
}

// f32 -> bf16 elementwise (n multiple of 4)
__global__ void cvt_bf16_kernel(const float* __restrict__ in,
                                bf16* __restrict__ out, int n)
{
    const int i = (blockIdx.x * blockDim.x + threadIdx.x) * 4;
    if (i >= n) return;
    const float4 v = *(const float4*)(in + i);
    out[i + 0] = (bf16)v.x;  out[i + 1] = (bf16)v.y;
    out[i + 2] = (bf16)v.z;  out[i + 3] = (bf16)v.w;
}

// W[K,Ncol] f32 -> Wt[Ncol,K] bf16
__global__ void transpose_w_kernel(const float* __restrict__ in,
                                   bf16* __restrict__ out, int K, int Ncol)
{
    const int t = blockIdx.x * blockDim.x + threadIdx.x;
    if (t >= K * Ncol) return;
    const int n = t / K, kk = t % K;
    out[t] = (bf16)in[(size_t)kk * Ncol + n];
}

// vb[M,P] bf16 -> vt[B,H,E,N] bf16
__global__ void transpose_v_kernel(const bf16* __restrict__ vb,
                                   bf16* __restrict__ vt)
{
    const int t = blockIdx.x * blockDim.x + threadIdx.x;  // B*H*E*N total
    const int s    = t & (N_ - 1);
    const int rest = t >> 10;
    const int e    = rest & (E_ - 1);
    const int bh   = rest >> 6;
    const int b = bh >> 3, h = bh & 7;
    vt[t] = vb[((size_t)(b * N_ + s)) * P_ + h * E_ + e];
}

// ---------------------------------------------------------------------------
// Host orchestration
// ---------------------------------------------------------------------------
extern "C" void kernel_launch(void* const* d_in, const int* in_sizes, int n_in,
                              void* d_out, int out_size, void* d_ws, size_t ws_size,
                              hipStream_t stream)
{
    const float* x    = (const float*)d_in[0];
    const float* Wq   = (const float*)d_in[1];
    const float* bq   = (const float*)d_in[2];
    const float* Wk   = (const float*)d_in[3];
    const float* bk   = (const float*)d_in[4];
    const float* Wv   = (const float*)d_in[5];
    const float* bv   = (const float*)d_in[6];
    const float* Wo   = (const float*)d_in[7];
    const float* bo   = (const float*)d_in[8];
    const float* ln1g = (const float*)d_in[9];
    const float* ln1b = (const float*)d_in[10];
    const float* W1   = (const float*)d_in[11];
    const float* b1   = (const float*)d_in[12];
    const float* W2   = (const float*)d_in[13];
    const float* b2   = (const float*)d_in[14];
    const float* ln2g = (const float*)d_in[15];
    const float* ln2b = (const float*)d_in[16];

    char* ws = (char*)d_ws;

    constexpr size_t SB   = (size_t)M_ * D_ * 2;   // 16 MB: bf16 [M,512]
    constexpr size_t SH1  = (size_t)M_ * F_ * 2;   // 67 MB: bf16 [M,2048]
    constexpr size_t SZW  = (size_t)D_ * P_ * 2;
    constexpr size_t SZW1 = (size_t)D_ * F_ * 2;

    const size_t o_xb   = 0;
    const size_t o_q    = 1 * SB;   // later reused as attpx/ffpx (f32, spans q+k)
    const size_t o_k    = 2 * SB;
    const size_t o_v    = 3 * SB;   // later reused as x1 (f32, spans v+vt)
    const size_t o_vt   = 4 * SB;
    const size_t o_vout = 5 * SB;   // later reused as x1 bf16
    const size_t o_h1   = 6 * SB;
    const size_t o_wq   = o_h1 + SH1;
    const size_t o_wk   = o_wq + SZW;
    const size_t o_wv   = o_wk + SZW;
    const size_t o_wo   = o_wv + SZW;
    const size_t o_w1   = o_wo + SZW;
    const size_t o_w2   = o_w1 + SZW1;

    bf16* xb   = (bf16*)(ws + o_xb);
    bf16* qb   = (bf16*)(ws + o_q);
    bf16* kb   = (bf16*)(ws + o_k);
    bf16* vb   = (bf16*)(ws + o_v);
    bf16* vtb  = (bf16*)(ws + o_vt);
    bf16* vout = (bf16*)(ws + o_vout);
    bf16* h1   = (bf16*)(ws + o_h1);
    bf16* wqt  = (bf16*)(ws + o_wq);
    bf16* wkt  = (bf16*)(ws + o_wk);
    bf16* wvt  = (bf16*)(ws + o_wv);
    bf16* wot  = (bf16*)(ws + o_wo);
    bf16* w1t  = (bf16*)(ws + o_w1);
    bf16* w2t  = (bf16*)(ws + o_w2);

    float* attpx = (float*)(ws + o_q);     // x + att        (aliases q,k)
    float* x1    = (float*)(ws + o_v);     // LN1 out f32    (aliases v,vt)
    bf16*  x1b   = (bf16*)(ws + o_vout);   // LN1 out bf16   (aliases vout)
    float* ffpx  = (float*)(ws + o_q);     // x1 + ffn       (aliases attpx)
    float* out   = (float*)d_out;

    // 1. convert x, transpose-convert weights
    cvt_bf16_kernel<<<(M_ * D_ / 4 + 255) / 256, 256, 0, stream>>>(x, xb, M_ * D_);
    transpose_w_kernel<<<(D_ * P_ + 255) / 256, 256, 0, stream>>>(Wq, wqt, D_, P_);
    transpose_w_kernel<<<(D_ * P_ + 255) / 256, 256, 0, stream>>>(Wk, wkt, D_, P_);
    transpose_w_kernel<<<(D_ * P_ + 255) / 256, 256, 0, stream>>>(Wv, wvt, D_, P_);
    transpose_w_kernel<<<(P_ * D_ + 255) / 256, 256, 0, stream>>>(Wo, wot, P_, D_);
    transpose_w_kernel<<<(D_ * F_ + 255) / 256, 256, 0, stream>>>(W1, w1t, D_, F_);
    transpose_w_kernel<<<(F_ * D_ + 255) / 256, 256, 0, stream>>>(W2, w2t, F_, D_);

    // 2. QKV projections
    dim3 gQKV(P_ / 64, M_ / 128);
    gemm_bf16_kernel<<<gQKV, 128, 0, stream>>>(xb, wqt, bq, nullptr, nullptr, qb, M_, P_, D_, 0);
    gemm_bf16_kernel<<<gQKV, 128, 0, stream>>>(xb, wkt, bk, nullptr, nullptr, kb, M_, P_, D_, 0);
    gemm_bf16_kernel<<<gQKV, 128, 0, stream>>>(xb, wvt, bv, nullptr, nullptr, vb, M_, P_, D_, 0);

    // 3. transpose V -> [B,H,E,N]
    transpose_v_kernel<<<(B_ * H_ * E_ * N_ + 255) / 256, 256, 0, stream>>>(vb, vtb);

    // 4. fused flash attention
    attn_kernel<<<B_ * H_ * (N_ / 64), 128, 0, stream>>>(qb, kb, vtb, vout);

    // 5. output projection + residual(x)
    dim3 gO(D_ / 64, M_ / 128);
    gemm_bf16_kernel<<<gO, 128, 0, stream>>>(vout, wot, bo, x, attpx, nullptr, M_, D_, P_, 0);

    // 6. LN1 -> x1 (f32) + x1 (bf16)
    ln_kernel<<<M_ / 8, 256, 0, stream>>>(attpx, ln1g, ln1b, x1, x1b);

    // 7. FFN up + exact GELU
    dim3 gF1(F_ / 64, M_ / 128);
    gemm_bf16_kernel<<<gF1, 128, 0, stream>>>(x1b, w1t, b1, nullptr, nullptr, h1, M_, F_, D_, 1);

    // 8. FFN down + residual(x1)
    gemm_bf16_kernel<<<gO, 128, 0, stream>>>(h1, w2t, b2, x1, ffpx, nullptr, M_, D_, F_, 0);

    // 9. LN2 -> output
    ln_kernel<<<M_ / 8, 256, 0, stream>>>(ffpx, ln2g, ln2b, out, nullptr);
}
